// SelfAttention_11373073399903
// MI455X (gfx1250) — matrix operand order repaired and checked by reading, not compiled, not run on hardware
//
#include <hip/hip_runtime.h>

typedef __attribute__((ext_vector_type(16))) __bf16          v16bf;
typedef __attribute__((ext_vector_type(8)))  float           v8f;
typedef __attribute__((ext_vector_type(8)))  unsigned short  ushortx8;
typedef __attribute__((ext_vector_type(4)))  unsigned int    uint4v;
typedef __attribute__((ext_vector_type(8)))  unsigned int    uint8v;

__device__ __forceinline__ unsigned short f32_to_bf16(float f) {
  unsigned int u = __float_as_uint(f);
  u += 0x7fffu + ((u >> 16) & 1u);          // round-to-nearest-even
  return (unsigned short)(u >> 16);
}

// A-fragment: 16x32 bf16, M rows at base (row stride in elements), K offset k0.
__device__ __forceinline__ v16bf load_a_frag(const unsigned short* base, int stride,
                                             int m0, int k0, int lane) {
  int r = m0 + (lane & 15);
  int h = (lane >> 4) << 3;                 // 0 or 8
  const unsigned short* p = base + (size_t)r * stride + k0;
  v16bf f;
  ushortx8* fp = (ushortx8*)&f;
  fp[0] = *(const ushortx8*)(p + h);
  fp[1] = *(const ushortx8*)(p + 16 + h);
  return f;
}

// B-fragment: 32x16, B[k][n] = src[n0+n][k]; src is N-major (contiguous K).
__device__ __forceinline__ v16bf load_b_frag(const unsigned short* base, int stride,
                                             int n0, int k0, int lane) {
  int r  = n0 + (lane & 15);
  int kb = k0 + ((lane >> 4) << 3);         // 0 or 8
  const unsigned short* p = base + (size_t)r * stride + kb;
  v16bf f;
  ushortx8* fp = (ushortx8*)&f;
  fp[0] = *(const ushortx8*)(p);
  fp[1] = *(const ushortx8*)(p + 16);
  return f;
}

// TDM: DMA one 2-D bf16 tile (tile_dim0 x tile_dim1 elems, row stride = tensor stride)
// from global into LDS, inserting 16B of pad per 1KB row (pad_interval=256dw, amount=4dw)
// so LDS rows are 1040B. Descriptor per CDNA5 ISA 8.3/8.4. Issued by one wave.
__device__ __forceinline__ void tdm_load_k_block(const unsigned short* gsrc,
                                                 unsigned int lds_off) {
  unsigned long long ga = (unsigned long long)(uintptr_t)gsrc;
  uint4v g0;
  g0[0] = 1u;                                       // count=1, user mode
  g0[1] = lds_off;                                  // lds_addr (bytes)
  g0[2] = (unsigned int)ga;                         // global_addr[31:0]
  g0[3] = (unsigned int)(ga >> 32) | (2u << 30);    // global_addr[56:32] | type=2
  uint8v g1;
  g1[0] = 0x07D10000u;   // data_size=2B | pad_enable | pad_interval=256dw | pad_amount=4dw
  g1[1] = 512u << 16;    // tensor_dim0 = 512  ([63:48])
  g1[2] = 2048u << 16;   // tensor_dim0 hi=0, tensor_dim1 = 2048 ([95:80])
  g1[3] = 512u << 16;    // tensor_dim1 hi=0, tile_dim0 = 512 ([127:112])
  g1[4] = 32u;           // tile_dim1 = 32, tile_dim2 = 0
  g1[5] = 512u;          // tensor_dim0_stride[31:0] = 512
  g1[6] = 0u;            // stride0 hi, stride1 lo
  g1[7] = 0u;            // stride1 hi
  asm volatile("tensor_load_to_lds %0, %1" : : "s"(g0), "s"(g1) : "memory");
}

// ---------------- conversion kernels ----------------
__global__ __launch_bounds__(256) void cvt_bf16(const float* __restrict__ src,
                                                unsigned short* __restrict__ dst, int n) {
  int i = blockIdx.x * blockDim.x + threadIdx.x;
  if (i < n) dst[i] = f32_to_bf16(src[i]);
}

// Wt[which][n][k] = W_which[k][n]  (bf16)
__global__ __launch_bounds__(256) void cvt_w_transpose(const float* __restrict__ Wq,
                                                       const float* __restrict__ Wk,
                                                       const float* __restrict__ Wv,
                                                       unsigned short* __restrict__ Wt) {
  int i = blockIdx.x * blockDim.x + threadIdx.x;     // 3 * 512 * 512 total
  int which = i >> 18;
  int rem = i & 262143;
  int n = rem >> 9;
  int k = rem & 511;
  const float* W = (which == 0) ? Wq : ((which == 1) ? Wk : Wv);
  Wt[i] = f32_to_bf16(W[(size_t)k * 512 + n]);
}

// ---------------- QKV projection GEMM ----------------
// Each wave computes a 16x64 C tile (4 accumulators, A-fragment reused x4).
__global__ __launch_bounds__(128) void qkv_gemm(const unsigned short* __restrict__ Xb,
                                                const unsigned short* __restrict__ Wt,
                                                const float* __restrict__ bq,
                                                const float* __restrict__ bk,
                                                const float* __restrict__ bv,
                                                unsigned short* __restrict__ Qb,
                                                unsigned short* __restrict__ Kb,
                                                unsigned short* __restrict__ Vtb) {
  const int lane  = threadIdx.x & 31;
  const int wave  = threadIdx.x >> 5;
  const int mtile = blockIdx.x * 4 + wave;   // 0..1023
  const int which = blockIdx.z;              // 0=Q 1=K 2=V
  const unsigned short* W = Wt + ((size_t)which << 18);
  const float* bias = (which == 0) ? bq : ((which == 1) ? bk : bv);

  const int m0 = mtile << 4;
  const int n0 = blockIdx.y << 6;            // 64-wide N tile
  v8f acc[4];
#pragma unroll
  for (int i = 0; i < 4; ++i) acc[i] = (v8f){};

#pragma unroll
  for (int k0 = 0; k0 < 512; k0 += 32) {
    v16bf a  = load_a_frag(Xb, 512, m0, k0, lane);
    v16bf b0 = load_b_frag(W, 512, n0,      k0, lane);
    v16bf b1 = load_b_frag(W, 512, n0 + 16, k0, lane);
    v16bf b2 = load_b_frag(W, 512, n0 + 32, k0, lane);
    v16bf b3 = load_b_frag(W, 512, n0 + 48, k0, lane);
    acc[0] = __builtin_amdgcn_wmma_f32_16x16x32_bf16(false, a, false, b0, (short)0, acc[0], false, false);
    acc[1] = __builtin_amdgcn_wmma_f32_16x16x32_bf16(false, a, false, b1, (short)0, acc[1], false, false);
    acc[2] = __builtin_amdgcn_wmma_f32_16x16x32_bf16(false, a, false, b2, (short)0, acc[2], false, false);
    acc[3] = __builtin_amdgcn_wmma_f32_16x16x32_bf16(false, a, false, b3, (short)0, acc[3], false, false);
  }

  const float scale = (which == 0) ? 0.04419417382415922f : 1.0f;  // 1/sqrt(512)
  const int rbase = m0 + ((lane >> 4) << 3);

#pragma unroll
  for (int nt = 0; nt < 4; ++nt) {
    const int ncol = n0 + nt * 16 + (lane & 15);
    const float bn = bias[ncol];
    if (which == 2) {
#pragma unroll
      for (int i = 0; i < 8; ++i) {
        int m = rbase + i;
        int bb = m >> 11, t = m & 2047;
        Vtb[((size_t)(bb * 512 + ncol) << 11) + t] = f32_to_bf16(acc[nt][i] + bn);
      }
    } else {
      unsigned short* O = (which == 0) ? Qb : Kb;
#pragma unroll
      for (int i = 0; i < 8; ++i) {
        O[(size_t)(rbase + i) * 512 + ncol] = f32_to_bf16((acc[nt][i] + bn) * scale);
      }
    }
  }
}

// ---------------- flash attention ----------------
// 1 block = 8 waves = (batch b, 32 query rows). wave = qsub*4 + es.
// K blocks are DMA'd into LDS by the Tensor Data Mover (wave 1), double-pumped
// against compute: issue for block i+1 during block i's softmax, wait before the
// end-of-iteration barrier; all K reads happen in phase 1, so one buffer suffices.
__global__ __launch_bounds__(256) void attn_kernel(const unsigned short* __restrict__ Qb,
                                                   const unsigned short* __restrict__ Kb,
                                                   const unsigned short* __restrict__ Vtb,
                                                   float* __restrict__ out) {
  __shared__ __align__(16) unsigned short Kt[32][520];     // TDM-staged K block (padded rows)
  __shared__ __align__(16) float part[2][4][16][32];       // partial scores [qsub][es]
  __shared__ __align__(16) unsigned short Pb[2][16][32];   // softmaxed probs (bf16)
  __shared__ __align__(16) float alphaL[2][16];
  __shared__ __align__(16) float rowl[2][16];

  const int lane = threadIdx.x & 31;
  const int wave = threadIdx.x >> 5;
  const int qsub = wave >> 2;                              // 0,1
  const int es   = wave & 3;                               // 0..3
  const int eoff = es << 7;                                // 0,128,256,384
  const int m0g  = blockIdx.y * 2048 + blockIdx.x * 32 + qsub * 16;

  const unsigned short* Kbase = Kb + (size_t)blockIdx.y * 2048 * 512;
  const unsigned short* Vbase = Vtb + ((size_t)blockIdx.y << 20);  // 512*2048 elems
  const unsigned int kt_off = (unsigned int)(unsigned long long)(uintptr_t)&Kt[0][0];

  // resident Q fragments for this wave's (qsub, E-slice)
  v16bf qf[4];
#pragma unroll
  for (int i = 0; i < 4; ++i) qf[i] = load_a_frag(Qb, 512, m0g, eoff + 32 * i, lane);

  v8f Oacc[8];
#pragma unroll
  for (int i = 0; i < 8; ++i) Oacc[i] = (v8f){};

  v8f mrow, lrow;
#pragma unroll
  for (int i = 0; i < 8; ++i) { mrow[i] = -3.0e38f; lrow[i] = 0.0f; }

  const int r = (lane >> 4) << 3;                          // D-layout row base (0 or 8)
  const int c = lane & 15;                                 // D-layout column

  // preload first K block via TDM
  if (wave == 1) {
    tdm_load_k_block(Kbase, kt_off);
    __builtin_amdgcn_s_wait_tensorcnt(0);
  }
  __syncthreads();

  for (int t0 = 0; t0 < 2048; t0 += 32) {
    const bool has_next = (t0 + 32) < 2048;

    // prefetch next V block (gfx1250 global_prefetch_b8)
    if (has_next)
      __builtin_prefetch(Vbase + (((size_t)(eoff + lane * 4)) << 11) + t0 + 32, 0, 3);

    // 1) partial scores over this wave's E-slice; K B-fragments from TDM-staged LDS
    v8f s0 = {}, s1 = {};
#pragma unroll
    for (int kk = 0; kk < 4; ++kk) {
      v16bf b0 = load_b_frag(&Kt[0][0], 520, 0,  eoff + 32 * kk, lane);
      v16bf b1 = load_b_frag(&Kt[0][0], 520, 16, eoff + 32 * kk, lane);
      s0 = __builtin_amdgcn_wmma_f32_16x16x32_bf16(false, qf[kk], false, b0, (short)0, s0, false, false);
      s1 = __builtin_amdgcn_wmma_f32_16x16x32_bf16(false, qf[kk], false, b1, (short)0, s1, false, false);
    }
#pragma unroll
    for (int i = 0; i < 8; ++i) {
      part[qsub][es][r + i][c]      = s0[i];
      part[qsub][es][r + i][c + 16] = s1[i];
    }
    __syncthreads();

    // 2a) wave 1: kick off the TDM for the next K block (overlaps softmax + PV)
    if (wave == 1 && has_next)
      tdm_load_k_block(Kbase + (size_t)(t0 + 32) * 512, kt_off);

    // 2b) waves 0 and 4: reduce partials, online softmax, publish P and alpha
    if (es == 0) {
      v8f f0, f1;
#pragma unroll
      for (int i = 0; i < 8; ++i) {
        f0[i] = part[qsub][0][r + i][c] + part[qsub][1][r + i][c] +
                part[qsub][2][r + i][c] + part[qsub][3][r + i][c];
        f1[i] = part[qsub][0][r + i][c + 16] + part[qsub][1][r + i][c + 16] +
                part[qsub][2][r + i][c + 16] + part[qsub][3][r + i][c + 16];
      }
      v8f mx;
#pragma unroll
      for (int i = 0; i < 8; ++i) mx[i] = fmaxf(f0[i], f1[i]);
#pragma unroll
      for (int d = 1; d < 16; d <<= 1) {
#pragma unroll
        for (int i = 0; i < 8; ++i) mx[i] = fmaxf(mx[i], __shfl_xor(mx[i], d, 32));
      }
      v8f mnew, alpha, p0, p1, sum;
#pragma unroll
      for (int i = 0; i < 8; ++i) {
        mnew[i]  = fmaxf(mrow[i], mx[i]);
        alpha[i] = __expf(mrow[i] - mnew[i]);
        p0[i]    = __expf(f0[i] - mnew[i]);
        p1[i]    = __expf(f1[i] - mnew[i]);
        sum[i]   = p0[i] + p1[i];
      }
#pragma unroll
      for (int d = 1; d < 16; d <<= 1) {
#pragma unroll
        for (int i = 0; i < 8; ++i) sum[i] += __shfl_xor(sum[i], d, 32);
      }
#pragma unroll
      for (int i = 0; i < 8; ++i) {
        lrow[i] = lrow[i] * alpha[i] + sum[i];
        mrow[i] = mnew[i];
        Pb[qsub][r + i][c]      = f32_to_bf16(p0[i]);
        Pb[qsub][r + i][c + 16] = f32_to_bf16(p1[i]);
      }
      if (c == 0) {
#pragma unroll
        for (int i = 0; i < 8; ++i) alphaL[qsub][r + i] = alpha[i];
      }
    }
    __syncthreads();

    // 3) all waves: rescale O, accumulate P x V^T over this key block
    {
      float al[8];
#pragma unroll
      for (int i = 0; i < 8; ++i) al[i] = alphaL[qsub][r + i];
      v16bf pf = load_a_frag(&Pb[qsub][0][0], 32, 0, 0, lane);
#pragma unroll
      for (int nt = 0; nt < 8; ++nt) {
        v16bf vf = load_b_frag(Vbase, 2048, eoff + nt * 16, t0, lane);
        v8f o = Oacc[nt];
#pragma unroll
        for (int i = 0; i < 8; ++i) o[i] *= al[i];
        Oacc[nt] = __builtin_amdgcn_wmma_f32_16x16x32_bf16(false, pf, false, vf, (short)0, o, false, false);
      }
    }
    // wave 1: make sure next K block landed in LDS before everyone loops around
    if (wave == 1 && has_next) __builtin_amdgcn_s_wait_tensorcnt(0);
    __syncthreads();
  }

  if (es == 0 && c == 0) {
#pragma unroll
    for (int i = 0; i < 8; ++i) rowl[qsub][r + i] = lrow[i];
  }
  __syncthreads();

  // normalize and store f32 output
  {
    float inv[8];
#pragma unroll
    for (int i = 0; i < 8; ++i) inv[i] = 1.0f / rowl[qsub][r + i];
    float* obase = out + (size_t)m0g * 512;
#pragma unroll
    for (int nt = 0; nt < 8; ++nt) {
      int col = eoff + nt * 16 + c;
#pragma unroll
      for (int i = 0; i < 8; ++i)
        obase[(size_t)(r + i) * 512 + col] = Oacc[nt][i] * inv[i];
    }
  }
}

extern "C" void kernel_launch(void* const* d_in, const int* in_sizes, int n_in,
                              void* d_out, int out_size, void* d_ws, size_t ws_size,
                              hipStream_t stream) {
  const float* x  = (const float*)d_in[0];
  const float* Wq = (const float*)d_in[1];
  const float* bq = (const float*)d_in[2];
  const float* Wk = (const float*)d_in[3];
  const float* bk = (const float*)d_in[4];
  const float* Wv = (const float*)d_in[5];
  const float* bv = (const float*)d_in[6];
  float* out = (float*)d_out;

  char* ws = (char*)d_ws;
  unsigned short* Xb  = (unsigned short*)(ws);                 // 16384x512 bf16 (16 MB)
  unsigned short* Wt  = (unsigned short*)(ws + (16ull << 20)); // 3x512x512 bf16 (1.5 MB)
  unsigned short* Qb  = (unsigned short*)(ws + (18ull << 20)); // 16 MB
  unsigned short* Kb  = (unsigned short*)(ws + (34ull << 20)); // 16 MB
  unsigned short* Vtb = (unsigned short*)(ws + (50ull << 20)); // [8][512][2048] 16 MB

  const int nx = 8 * 2048 * 512;
  cvt_bf16<<<(nx + 255) / 256, 256, 0, stream>>>(x, Xb, nx);
  cvt_w_transpose<<<(3 * 512 * 512) / 256, 256, 0, stream>>>(Wq, Wk, Wv, Wt);

  dim3 g1(256, 8, 3);    // 4 wave-tiles/block x 256 = 1024 M-tiles; 8 x 64-wide N; Q/K/V
  qkv_gemm<<<g1, 128, 0, stream>>>(Xb, Wt, bq, bk, bv, Qb, Kb, Vtb);

  dim3 g2(64, 8);        // S/32 query blocks x B
  attn_kernel<<<g2, 256, 0, stream>>>(Qb, Kb, Vtb, out);
}
